// TransformerLayer_32710470926925
// MI455X (gfx1250) — matrix operand order repaired and checked by reading, not compiled, not run on hardware
//
#include <hip/hip_runtime.h>
#include <hip/hip_bf16.h>

// Problem sizes (fixed by reference)
constexpr int kB = 8;
constexpr int kC = 128;
constexpr int kN = 4096;

typedef __bf16 bf16_t;
typedef __attribute__((ext_vector_type(16))) __bf16 v16bf;
typedef __attribute__((ext_vector_type(8)))  __bf16 v8bf;
typedef __attribute__((ext_vector_type(8)))  float  v8f;

// ---------------------------------------------------------------------------
// WMMA helper: D(f32 16x16) = A(bf16 16x32) * B(bf16 32x16) + C
// ---------------------------------------------------------------------------
static __device__ __forceinline__ v8f wmma_bf16(v16bf a, v16bf b, v8f c) {
  return __builtin_amdgcn_wmma_f32_16x16x32_bf16(
      /*neg_a=*/false, a, /*neg_b=*/false, b,
      /*c_mod=*/(short)0, c, /*reuse_a=*/false, /*reuse_b=*/false);
}

// A-matrix 16x32 bf16 load (ISA 7.12.2): lane<16 holds row (lane), K {0..7,16..23};
// lane>=16 holds row (lane-16), K {8..15,24..31}. base = src + row0*ld + k0.
static __device__ __forceinline__ v16bf load_a_bf16(const bf16_t* base, int ld, int lane) {
  int r = lane & 15, half = lane >> 4;
  const bf16_t* p = base + (size_t)r * ld;
  v8bf lo = *(const v8bf*)(p + half * 8);
  v8bf hi = *(const v8bf*)(p + 16 + half * 8);
  v16bf a;
#pragma unroll
  for (int i = 0; i < 8; ++i) { a[i] = lo[i]; a[8 + i] = hi[i]; }
  return a;
}

// Same A layout, but source is fp32 (weights), converted to bf16 on the fly.
static __device__ __forceinline__ v16bf load_a_f32(const float* base, int ld, int lane) {
  int r = lane & 15, half = lane >> 4;
  const float* p = base + (size_t)r * ld;
  v8f lo = *(const v8f*)(p + half * 8);
  v8f hi = *(const v8f*)(p + 16 + half * 8);
  v16bf a;
#pragma unroll
  for (int i = 0; i < 8; ++i) { a[i] = (bf16_t)lo[i]; a[8 + i] = (bf16_t)hi[i]; }
  return a;
}

// B-matrix 32x16 bf16 load: lane<16 holds column (lane), K {0..7,16..23}; lane>=16 holds
// column (lane-16), K {8..15,24..31}. Storage must have K contiguous at fixed column:
// base = src + col0*ld + k0, element addr = base + col*ld + half*8 + (j<8 ? j : 8+j).
static __device__ __forceinline__ v16bf load_b_bf16(const bf16_t* base, int ld, int lane) {
  int c = lane & 15, half = lane >> 4;
  const bf16_t* p = base + (size_t)c * ld + half * 8;
  v8bf lo = *(const v8bf*)(p);
  v8bf hi = *(const v8bf*)(p + 16);
  v16bf b;
#pragma unroll
  for (int i = 0; i < 8; ++i) { b[i] = lo[i]; b[8 + i] = hi[i]; }
  return b;
}

// Same B layout from fp32 source (weights, row-major W[col][k] contiguous in k).
static __device__ __forceinline__ v16bf load_b_f32(const float* base, int ld, int lane) {
  int c = lane & 15, half = lane >> 4;
  const float* p = base + (size_t)c * ld + half * 8;
  v8f lo = *(const v8f*)(p);
  v8f hi = *(const v8f*)(p + 16);
  v16bf b;
#pragma unroll
  for (int i = 0; i < 8; ++i) { b[i] = (bf16_t)lo[i]; b[8 + i] = (bf16_t)hi[i]; }
  return b;
}

// ---------------------------------------------------------------------------
// Kernel 0: x[B,C,N] f32 -> xT[B,N,C] bf16 (coalesced LDS transpose)
// ---------------------------------------------------------------------------
__global__ __launch_bounds__(256) void k_transpose(const float* __restrict__ x,
                                                   bf16_t* __restrict__ xT) {
  __shared__ float tile[32][33];
  int b = blockIdx.z, c0 = blockIdx.y * 32, n0 = blockIdx.x * 32;
  int tx = threadIdx.x, ty = threadIdx.y;  // 32 x 8
  const float* xb = x + (size_t)b * kC * kN;
  bf16_t* xTb = xT + (size_t)b * kN * kC;
#pragma unroll
  for (int i = 0; i < 4; ++i) {
    int row = ty + i * 8;
    tile[row][tx] = xb[(size_t)(c0 + row) * kN + n0 + tx];
  }
  __syncthreads();
#pragma unroll
  for (int i = 0; i < 4; ++i) {
    int row = ty + i * 8;
    xTb[(size_t)(n0 + row) * kC + c0 + tx] = (bf16_t)tile[tx][row];
  }
}

// ---------------------------------------------------------------------------
// Kernel 1: y = W @ x + b, store transposed: yT[B,N,C] bf16 (used for q and k)
// GEMM orientation: M = c_out tile, Ncols = n tile, K = c_in (128).
// One wave per 16x16 tile; packed 16B stores.
// ---------------------------------------------------------------------------
__global__ __launch_bounds__(256) void k_proj_qk(const bf16_t* __restrict__ xT,
                                                 const float* __restrict__ W,
                                                 const float* __restrict__ bias,
                                                 bf16_t* __restrict__ yT) {
  int lane = threadIdx.x & 31;
  int wave = threadIdx.x >> 5;
  int tile = blockIdx.x * 8 + wave;
  int b  = tile >> 11;         // 2048 tiles per batch
  int ct = (tile >> 8) & 7;    // c_out tile (8)
  int nt = tile & 255;         // n tile (256)
  const bf16_t* xTb = xT + (size_t)b * kN * kC;

  v8f acc = {};
#pragma unroll
  for (int kk = 0; kk < 4; ++kk) {
    v16bf a  = load_a_f32(W + (size_t)ct * 16 * kC + kk * 32, kC, lane);
    v16bf bm = load_b_bf16(xTb + (size_t)nt * 16 * kC + kk * 32, kC, lane);
    acc = wmma_bf16(a, bm, acc);
  }
  int half = lane >> 4;
  v8f bv = *(const v8f*)(bias + ct * 16 + half * 8);  // c_out = ct*16 + half*8 + r
  int n = nt * 16 + (lane & 15);
  bf16_t* dst = yT + (size_t)b * kN * kC + (size_t)n * kC + ct * 16 + half * 8;
  v8bf o;
#pragma unroll
  for (int r = 0; r < 8; ++r) o[r] = (bf16_t)(acc[r] + bv[r]);
  *(v8bf*)dst = o;
}

// ---------------------------------------------------------------------------
// Kernel 2: y = W @ x + b, store natural layout y[B,C,N] bf16 (used for v)
// GEMM orientation: M = n tile, Ncols = c_out tile, K = c_in.
// ---------------------------------------------------------------------------
__global__ __launch_bounds__(256) void k_proj_v(const bf16_t* __restrict__ xT,
                                                const float* __restrict__ W,
                                                const float* __restrict__ bias,
                                                bf16_t* __restrict__ y) {
  int lane = threadIdx.x & 31;
  int wave = threadIdx.x >> 5;
  int tile = blockIdx.x * 8 + wave;
  int b  = tile >> 11;
  int ct = (tile >> 8) & 7;
  int nt = tile & 255;
  const bf16_t* xTb = xT + (size_t)b * kN * kC;

  v8f acc = {};
#pragma unroll
  for (int kk = 0; kk < 4; ++kk) {
    v16bf a  = load_a_bf16(xTb + (size_t)nt * 16 * kC + kk * 32, kC, lane);
    v16bf bm = load_b_f32(W + (size_t)ct * 16 * kC + kk * 32, kC, lane);
    acc = wmma_bf16(a, bm, acc);
  }
  int half = lane >> 4;
  int c = ct * 16 + (lane & 15);
  float bs = bias[c];
  bf16_t* dst = y + (size_t)b * kC * kN + (size_t)c * kN + nt * 16 + half * 8;
  v8bf o;
#pragma unroll
  for (int r = 0; r < 8; ++r) o[r] = (bf16_t)(acc[r] + bs);
  *(v8bf*)dst = o;
}

// ---------------------------------------------------------------------------
// Kernel 3: flash attention. One wave per 16-query tile; 4096 keys processed
// in chunks of 64 (4 score tiles, 2 P A-tiles, 16 PV WMMAs -> 32 WMMAs per
// softmax pass). Online softmax: each D-layout row lives in one 16-lane half,
// so butterfly shfl_xor masks 1..8 reduce across the row's 16 columns.
// P (f32 D-layout) -> bf16 A-layout goes through a PER-WAVE private LDS
// region; same-wave LDS ops are executed in order by the LDS unit (ISA §7.3),
// so only a compiler-level fence (wave_barrier) is needed, not s_barrier.
// ---------------------------------------------------------------------------
__global__ __launch_bounds__(256) void k_flash(const bf16_t* __restrict__ qT,
                                               const bf16_t* __restrict__ kT,
                                               const bf16_t* __restrict__ v,
                                               bf16_t* __restrict__ aT) {
  __shared__ bf16_t Plds[8][16][64];  // per-wave P tile (16 x 64 keys), 16KB
  int lane = threadIdx.x & 31;
  int wave = threadIdx.x >> 5;
  int tile = blockIdx.x * 8 + wave;
  int b  = tile >> 8;     // 256 query tiles per batch
  int it = tile & 255;
  int half = lane >> 4;
  int col  = lane & 15;

  const bf16_t* qTb = qT + (size_t)b * kN * kC;
  const bf16_t* kTb = kT + (size_t)b * kN * kC;
  const bf16_t* vb  = v  + (size_t)b * kC * kN;
  bf16_t* aTb = aT + (size_t)b * kN * kC;

  // Resident q A-tiles covering K = 0..127
  v16bf qa[4];
#pragma unroll
  for (int kk = 0; kk < 4; ++kk)
    qa[kk] = load_a_bf16(qTb + (size_t)it * 16 * kC + kk * 32, kC, lane);

  v8f O[8] = {};          // 16 rows x 128 channels accumulator (8 c-tiles)
  float m[8], l[8];
#pragma unroll
  for (int e = 0; e < 8; ++e) { m[e] = -1e30f; l[e] = 0.0f; }

  bf16_t(*P)[64] = Plds[wave];

  for (int jc = 0; jc < 64; ++jc) {
    int j0 = jc * 64;
    if (jc + 1 < 64) {  // lane-spread prefetch: 2 insts cover 64 rows (16KB)
      __builtin_prefetch(kTb + (size_t)(j0 + 64 + lane) * kC, 0, 0);
      __builtin_prefetch(kTb + (size_t)(j0 + 96 + lane) * kC, 0, 0);
    }
    // ---- scores: S = q^T k for 64 keys (four 16x16 tiles) ----
    v8f S[4] = {};
#pragma unroll
    for (int kk = 0; kk < 4; ++kk) {
#pragma unroll
      for (int t = 0; t < 4; ++t) {
        v16bf bk = load_b_bf16(kTb + (size_t)(j0 + t * 16) * kC + kk * 32, kC, lane);
        S[t] = wmma_bf16(qa[kk], bk, S[t]);
      }
    }
    // ---- online softmax (row = e + 8*half, columns across 16-lane half) ----
    float scale[8];
#pragma unroll
    for (int e = 0; e < 8; ++e) {
      float t01 = fmaxf(S[0][e], S[1][e]);
      float t23 = fmaxf(S[2][e], S[3][e]);
      float t = fmaxf(t01, t23);
#pragma unroll
      for (int mk = 1; mk < 16; mk <<= 1) t = fmaxf(t, __shfl_xor(t, mk, 32));
      float mnew = fmaxf(m[e], t);
      float sc = __expf(m[e] - mnew);
      float p0 = __expf(S[0][e] - mnew);
      float p1 = __expf(S[1][e] - mnew);
      float p2 = __expf(S[2][e] - mnew);
      float p3 = __expf(S[3][e] - mnew);
      float rs = (p0 + p1) + (p2 + p3);
#pragma unroll
      for (int mk = 1; mk < 16; mk <<= 1) rs += __shfl_xor(rs, mk, 32);
      l[e] = l[e] * sc + rs;
      m[e] = mnew;
      scale[e] = sc;
      S[0][e] = p0; S[1][e] = p1; S[2][e] = p2; S[3][e] = p3;
    }
#pragma unroll
    for (int t = 0; t < 8; ++t)
#pragma unroll
      for (int e = 0; e < 8; ++e) O[t][e] *= scale[e];

    // ---- P: D-layout f32 -> A-layout bf16 via per-wave LDS round trip ----
    __builtin_amdgcn_wave_barrier();   // WAR fence vs previous iter's loads
#pragma unroll
    for (int t = 0; t < 4; ++t)
#pragma unroll
      for (int e = 0; e < 8; ++e)
        P[e + 8 * half][t * 16 + col] = (bf16_t)S[t][e];
    __builtin_amdgcn_wave_barrier();   // store->load ordering (same-wave LDS
                                       // ops complete in order at the LDS unit)
    v16bf pa0 = load_a_bf16(&P[0][0],      64, lane);  // keys j0+ 0..31
    v16bf pa1 = load_a_bf16(&P[0][0] + 32, 64, lane);  // keys j0+32..63

    // ---- O += P @ V for all 8 channel tiles (K = 64 keys, 2 WMMAs each) ----
#pragma unroll
    for (int t = 0; t < 8; ++t) {
      v16bf bv0 = load_b_bf16(vb + (size_t)t * 16 * kN + j0, kN, lane);
      O[t] = wmma_bf16(pa0, bv0, O[t]);
      v16bf bv1 = load_b_bf16(vb + (size_t)t * 16 * kN + j0 + 32, kN, lane);
      O[t] = wmma_bf16(pa1, bv1, O[t]);
    }
  }

  // ---- normalize and store attn_out as aT[B,N,C] bf16 ----
  float inv[8];
#pragma unroll
  for (int e = 0; e < 8; ++e) inv[e] = 1.0f / l[e];
#pragma unroll
  for (int t = 0; t < 8; ++t) {
#pragma unroll
    for (int e = 0; e < 8; ++e) {
      aTb[(size_t)(it * 16 + e + 8 * half) * kC + t * 16 + col] =
          (bf16_t)(O[t][e] * inv[e]);
    }
  }
}

// ---------------------------------------------------------------------------
// Kernel 4: out = Wf @ attn_out + bf + x, fp32 out[B,C,N].
// Same orientation as k_proj_v; 32B vector stores of the fp32 result.
// ---------------------------------------------------------------------------
__global__ __launch_bounds__(256) void k_final(const bf16_t* __restrict__ aT,
                                               const float* __restrict__ W,
                                               const float* __restrict__ bias,
                                               const float* __restrict__ x,
                                               float* __restrict__ out) {
  int lane = threadIdx.x & 31;
  int wave = threadIdx.x >> 5;
  int tile = blockIdx.x * 8 + wave;
  int b  = tile >> 11;
  int ct = (tile >> 8) & 7;
  int nt = tile & 255;
  const bf16_t* aTb = aT + (size_t)b * kN * kC;

  v8f acc = {};
#pragma unroll
  for (int kk = 0; kk < 4; ++kk) {
    v16bf a  = load_a_bf16(aTb + (size_t)nt * 16 * kC + kk * 32, kC, lane);
    v16bf bm = load_b_f32(W + (size_t)ct * 16 * kC + kk * 32, kC, lane);
    acc = wmma_bf16(a, bm, acc);
  }
  int half = lane >> 4;
  int c = ct * 16 + (lane & 15);
  float bs = bias[c];
  size_t off = (size_t)b * kC * kN + (size_t)c * kN + nt * 16 + half * 8;
  v8f xv = *(const v8f*)(x + off);
  v8f o;
#pragma unroll
  for (int r = 0; r < 8; ++r) o[r] = acc[r] + bs + xv[r];
  *(v8f*)(out + off) = o;
}

// ---------------------------------------------------------------------------
// Host launcher
// ---------------------------------------------------------------------------
extern "C" void kernel_launch(void* const* d_in, const int* in_sizes, int n_in,
                              void* d_out, int out_size, void* d_ws, size_t ws_size,
                              hipStream_t stream) {
  (void)in_sizes; (void)n_in; (void)out_size; (void)ws_size;
  const float* x  = (const float*)d_in[0];
  const float* Wq = (const float*)d_in[1];
  const float* bq = (const float*)d_in[2];
  const float* Wk = (const float*)d_in[3];
  const float* bk = (const float*)d_in[4];
  const float* Wv = (const float*)d_in[5];
  const float* bv = (const float*)d_in[6];
  const float* Wf = (const float*)d_in[7];
  const float* bf = (const float*)d_in[8];
  float* out = (float*)d_out;

  const size_t SZ = (size_t)kB * kN * kC;  // 4M elements per tensor
  bf16_t* xT = (bf16_t*)d_ws;   // [B,N,C]
  bf16_t* qT = xT + SZ;         // [B,N,C]
  bf16_t* kT = qT + SZ;         // [B,N,C]
  bf16_t* vv = kT + SZ;         // [B,C,N]
  bf16_t* aT = vv + SZ;         // [B,N,C]   (total 40 MB workspace)

  k_transpose<<<dim3(kN / 32, kC / 32, kB), dim3(32, 8), 0, stream>>>(x, xT);
  k_proj_qk<<<2048, 256, 0, stream>>>(xT, Wq, bq, qT);
  k_proj_qk<<<2048, 256, 0, stream>>>(xT, Wk, bk, kT);
  k_proj_v <<<2048, 256, 0, stream>>>(xT, Wv, bv, vv);
  k_flash  <<<256,  256, 0, stream>>>(qT, kT, vv, aT);
  k_final  <<<2048, 256, 0, stream>>>(aT, Wf, bf, x, out);
}
